// RelPositionMultiheadAttentionWeights_60352880443553
// MI455X (gfx1250) — matrix operand order repaired and checked by reading, not compiled, not run on hardware
//
#include <hip/hip_runtime.h>

// ---------------------------------------------------------------------------
// RelPositionMultiheadAttentionWeights for MI455X (gfx1250, wave32, WMMA)
//
//   1) pe_kernel     : pe[h,t,d] = dot(pos_emb[t,:], W_pos[h*4+d,:])   (tiny)
//   2) cvt_kernel    : x, W_in  -> f16 copies (WMMA operands)
//   3) inproj_kernel : WMMA GEMM (8192 x 544 x 512), scatter into
//                      qh/kh (f16, [b*8+h][s][32]) and pf (f32, [b*8+h][s][4])
//   4) attn_kernel   : per (h,b) WMMA QK^T + rel-pos shift + mask + online
//                      softmax. K slice / pe window / mask row staged in LDS
//                      once per block (async-to-LDS when available); hot loop
//                      is WMMA + ds_load only; 134 MB output written once.
// ---------------------------------------------------------------------------

#define S_LEN   512
#define BATCH   16
#define EMB     512
#define HEADS   8
#define DQ      32
#define DP      4
#define NPROJ   544                 // (32+32+4)*8
#define RELLEN  (2*S_LEN - 1)       // 1023
#define MROWS   (S_LEN*BATCH)       // 8192
#define ROWS_PER_BLK 128            // attn: 8 waves x 16 rows
#define PE_WIN  (S_LEN + ROWS_PER_BLK - 1)   // 639 rel entries per block

typedef _Float16 v16h __attribute__((ext_vector_type(16)));
typedef _Float16 v8h  __attribute__((ext_vector_type(8)));
typedef float    v8f  __attribute__((ext_vector_type(8)));
typedef float    v4f  __attribute__((ext_vector_type(4)));
typedef int      v4i  __attribute__((ext_vector_type(4)));

// ---------------------------------------------------------------------------
// CDNA5 async global->LDS staging (GLOBAL_LOAD_ASYNC_TO_LDS_B128, ASYNCcnt).
// Builtin signature (from hipcc diagnostics): (int4 AS1*, int4 AS3*, imm, imm).
// Gated on builtin availability; falls back to a synchronous vector copy.
// ---------------------------------------------------------------------------
#if defined(__HIP_DEVICE_COMPILE__)
  #if __has_builtin(__builtin_amdgcn_global_load_async_to_lds_b128)
    #define ASYNC_LDS 1
  #endif
#endif

__device__ __forceinline__ void stage16(void* lds_dst, const void* gsrc) {
#if defined(ASYNC_LDS)
    __builtin_amdgcn_global_load_async_to_lds_b128(
        (__attribute__((address_space(1))) v4i*)gsrc,
        (__attribute__((address_space(3))) v4i*)lds_dst,
        0, 0);
#else
    *(v8h*)lds_dst = *(const v8h*)gsrc;
#endif
}

__device__ __forceinline__ void stage_drain() {
#if defined(ASYNC_LDS)
  #if __has_builtin(__builtin_amdgcn_s_wait_asynccnt)
    __builtin_amdgcn_s_wait_asynccnt(0);
  #else
    asm volatile("s_wait_asynccnt 0x0" ::: "memory");
  #endif
#endif
}

// A-fragment (16x32 f16, M x K): lane = row (lane%16);
// v16h elems 0..7  <- K = k0 + 8*half .. +8
// v16h elems 8..15 <- K = k0 + 16 + 8*half .. +8      (ISA 7.12.2)
__device__ __forceinline__ v16h load_a_frag(const _Float16* row, int k0, int half) {
    v8h lo = *(const v8h*)(row + k0 + 8*half);
    v8h hi = *(const v8h*)(row + k0 + 16 + 8*half);
    return __builtin_shufflevector(lo, hi, 0,1,2,3,4,5,6,7,8,9,10,11,12,13,14,15);
}

// B-fragment (32x16 f16, K x N) loaded from row-major B^T (N x K):
// lane = col (lane%16); v16h elems 0..7 <- K = k0 + 8*half .. +8, elems 8..15 <- K = k0 + 16 + 8*half .. +8.
__device__ __forceinline__ v16h load_b_frag(const _Float16* rowT, int k0, int half) {
    v8h lo = *(const v8h*)(rowT + k0 + 8*half);
    v8h hi = *(const v8h*)(rowT + k0 + 16 + 8*half);
    return __builtin_shufflevector(lo, hi, 0,1,2,3,4,5,6,7,8,9,10,11,12,13,14,15);
}

// ---------------------------------------------------------------------------
__global__ __launch_bounds__(32) void pe_kernel(const float* __restrict__ pos_emb,
                                                const float* __restrict__ wpos,
                                                float* __restrict__ pe) {
    const int t   = blockIdx.x;        // 0..1022
    const int tid = threadIdx.x;       // 0..31 == h*4+d
    const float* pr = pos_emb + (size_t)t * 192;
    const float* wr = wpos    + (size_t)tid * 192;
    float acc = 0.f;
    #pragma unroll 8
    for (int p = 0; p < 192; ++p) acc = fmaf(pr[p], wr[p], acc);
    const int h = tid >> 2, d = tid & 3;
    pe[((size_t)h * RELLEN + t) * DP + d] = acc;   // (H, 1023, 4) -> float4 rows
}

// ---------------------------------------------------------------------------
__global__ __launch_bounds__(256) void cvt_kernel(const float* __restrict__ x,
                                                  const float* __restrict__ w,
                                                  _Float16* __restrict__ xh,
                                                  _Float16* __restrict__ wh) {
    const size_t NX = (size_t)MROWS * EMB;
    const size_t NW = (size_t)NPROJ * EMB;
    size_t i = (size_t)blockIdx.x * 256 + threadIdx.x;
    if (i < NX)            xh[i]      = (_Float16)x[i];
    else if (i < NX + NW)  wh[i - NX] = (_Float16)w[i - NX];
}

// ---------------------------------------------------------------------------
__global__ __launch_bounds__(128) void inproj_kernel(const _Float16* __restrict__ xh,
                                                     const _Float16* __restrict__ wh,
                                                     const float* __restrict__ bias,
                                                     _Float16* __restrict__ qh,
                                                     _Float16* __restrict__ kh,
                                                     float* __restrict__ pf) {
    const int lane = threadIdx.x & 31;
    const int wave = threadIdx.x >> 5;
    const int half = lane >> 4;
    const int nl   = lane & 15;
    const int m0   = (blockIdx.x * 4 + wave) * 16;
    const int n0   = blockIdx.y * 16;

    const _Float16* xr = xh + (size_t)(m0 + nl) * EMB;   // A rows
    const _Float16* wr = wh + (size_t)(n0 + nl) * EMB;   // B^T rows (= W rows)

    v8f acc = {};
    #pragma unroll
    for (int kk = 0; kk < EMB / 32; ++kk) {
        v16h a = load_a_frag(xr, kk * 32, half);
        v16h b = load_b_frag(wr, kk * 32, half);
        acc = __builtin_amdgcn_wmma_f32_16x16x32_f16(false, a, false, b,
                                                     (short)0, acc, false, false);
    }

    const int o  = n0 + nl;
    const float bv = bias[o];
    #pragma unroll
    for (int v = 0; v < 8; ++v) {
        const int r  = m0 + v + 8 * half;   // flat (s*B + b) row
        const int s  = r >> 4;              // B == 16
        const int bq = r & 15;
        const float val = acc[v] + bv;
        if (o < 256) {                      // q
            const int h = o >> 5, d = o & 31;
            qh[((size_t)(bq * HEADS + h) * S_LEN + s) * DQ + d] = (_Float16)val;
        } else if (o < 512) {               // k
            const int o2 = o - 256, h = o2 >> 5, d = o2 & 31;
            kh[((size_t)(bq * HEADS + h) * S_LEN + s) * DQ + d] = (_Float16)val;
        } else {                            // p (keep f32: only 4 dims)
            const int o2 = o - 512, h = o2 >> 2, dp = o2 & 3;
            pf[((size_t)(bq * HEADS + h) * S_LEN + s) * DP + dp] = val;
        }
    }
}

// ---------------------------------------------------------------------------
__global__ __launch_bounds__(256) void attn_kernel(const _Float16* __restrict__ qh,
                                                   const _Float16* __restrict__ kh,
                                                   const float* __restrict__ pf,
                                                   const float* __restrict__ pe,
                                                   const unsigned char* __restrict__ mask,
                                                   float* __restrict__ out) {
    __shared__ _Float16      ksh[S_LEN * DQ];        // 32768 B
    __shared__ float         pesh[PE_WIN * DP];      // 10224 B
    __shared__ unsigned char msh[S_LEN];             //   512 B

    const int tid  = threadIdx.x;
    const int lane = tid & 31;
    const int wave = tid >> 5;
    const int half = lane >> 4;
    const int nl   = lane & 15;
    const int bh   = blockIdx.y;            // b*8 + h
    const int b    = bh >> 3, h = bh & 7;
    const int I0   = blockIdx.x * ROWS_PER_BLK;
    const int i0   = I0 + wave * 16;

    // ---- stage K slice (512x32 f16), pe window, mask row into LDS --------
    const _Float16* gk = kh + (size_t)bh * S_LEN * DQ;     // 2048 x 16B chunks
    #pragma unroll
    for (int c = 0; c < 8; ++c) {
        const int ch = tid + 256 * c;
        stage16(ksh + ch * 8, gk + ch * 8);
    }
    // block touches rel = j - i + 511, i in [I0,I0+128), j in [0,512)
    //   -> rel in [384-I0, 1022-I0], width 639; rel_local = rel - (384-I0)
    const float* gpe = pe + ((size_t)h * RELLEN + (384 - I0)) * DP;
    for (int e = tid; e < PE_WIN; e += 256)
        stage16(pesh + e * 4, gpe + (size_t)e * 4);
    if (tid < S_LEN / 4)
        *(unsigned int*)(msh + tid * 4) =
            *(const unsigned int*)(mask + b * S_LEN + tid * 4);

    // Q fragment + p rows (global, once per wave) overlap the async staging
    const _Float16* qrow = qh + (size_t)(bh * S_LEN + i0 + nl) * DQ;
    const v16h a = load_a_frag(qrow, 0, half);
    const float* pfh = pf + (size_t)bh * S_LEN * DP;
    v4f prow[8];
    #pragma unroll
    for (int v = 0; v < 8; ++v)
        prow[v] = *(const v4f*)(pfh + (size_t)(i0 + v + 8 * half) * DP);

    stage_drain();          // s_wait_asynccnt 0 (wave's own async copies)
    __syncthreads();        // make every wave's staging visible block-wide

    float m[8], l[8];
    #pragma unroll
    for (int v = 0; v < 8; ++v) { m[v] = -3.0e38f; l[v] = 0.f; }

    const int il0 = wave * 16 + 8 * half;    // local row base (+v below)

    // ---- pass 1: online row max + sum of exp -----------------------------
    for (int jt = 0; jt < S_LEN / 16; ++jt) {
        const int j0 = jt * 16;
        const v16h bf = load_b_frag(ksh + (size_t)(j0 + nl) * DQ, 0, half);
        v8f c = {};
        c = __builtin_amdgcn_wmma_f32_16x16x32_f16(false, a, false, bf,
                                                   (short)0, c, false, false);
        const int  j   = j0 + nl;
        const bool msk = msh[j] != 0;
        #pragma unroll
        for (int v = 0; v < 8; ++v) {
            const int rl = j - (il0 + v) + 127;           // rel_local [0,638]
            const v4f pev = *(const v4f*)(pesh + (size_t)rl * DP);
            float s = c[v] + prow[v].x * pev.x + prow[v].y * pev.y
                           + prow[v].z * pev.z + prow[v].w * pev.w;
            if (msk) s = -1000.f;
            float tm = s;                       // 16-lane group reductions
            tm = fmaxf(tm, __shfl_xor(tm, 1, 32));
            tm = fmaxf(tm, __shfl_xor(tm, 2, 32));
            tm = fmaxf(tm, __shfl_xor(tm, 4, 32));
            tm = fmaxf(tm, __shfl_xor(tm, 8, 32));
            const float mn = fmaxf(m[v], tm);
            float e = __expf(s - mn);
            e += __shfl_xor(e, 1, 32);
            e += __shfl_xor(e, 2, 32);
            e += __shfl_xor(e, 4, 32);
            e += __shfl_xor(e, 8, 32);
            l[v] = l[v] * __expf(m[v] - mn) + e;
            m[v] = mn;
        }
    }

    float inv[8];
    #pragma unroll
    for (int v = 0; v < 8; ++v) inv[v] = 1.0f / l[v];

    // ---- pass 2: recompute scores (WMMA ~free), write probabilities ------
    for (int jt = 0; jt < S_LEN / 16; ++jt) {
        const int j0 = jt * 16;
        const v16h bf = load_b_frag(ksh + (size_t)(j0 + nl) * DQ, 0, half);
        v8f c = {};
        c = __builtin_amdgcn_wmma_f32_16x16x32_f16(false, a, false, bf,
                                                   (short)0, c, false, false);
        const int  j   = j0 + nl;
        const bool msk = msh[j] != 0;
        #pragma unroll
        for (int v = 0; v < 8; ++v) {
            const int i  = i0 + v + 8 * half;
            const int rl = j - (il0 + v) + 127;
            const v4f pev = *(const v4f*)(pesh + (size_t)rl * DP);
            float s = c[v] + prow[v].x * pev.x + prow[v].y * pev.y
                           + prow[v].z * pev.z + prow[v].w * pev.w;
            if (msk) s = -1000.f;
            out[(((size_t)h * BATCH + b) * S_LEN + i) * S_LEN + j] =
                __expf(s - m[v]) * inv[v];      // out layout (H,B,S,S)
        }
    }
}

// ---------------------------------------------------------------------------
// Workspace layout (256B-aligned slots, ~18.5 MB total)
// ---------------------------------------------------------------------------
static constexpr size_t SZ_XH = (size_t)MROWS * EMB * 2;
static constexpr size_t SZ_WH = (size_t)NPROJ * EMB * 2;
static constexpr size_t SZ_QH = (size_t)BATCH * HEADS * S_LEN * DQ * 2;
static constexpr size_t SZ_PF = (size_t)BATCH * HEADS * S_LEN * DP * 4;
static constexpr size_t SZ_PE = (size_t)HEADS * RELLEN * DP * 4;
static constexpr size_t A256(size_t v) { return (v + 255) & ~(size_t)255; }
static constexpr size_t OFF_XH = 0;
static constexpr size_t OFF_WH = OFF_XH + A256(SZ_XH);
static constexpr size_t OFF_QH = OFF_WH + A256(SZ_WH);
static constexpr size_t OFF_KH = OFF_QH + A256(SZ_QH);
static constexpr size_t OFF_PF = OFF_KH + A256(SZ_QH);
static constexpr size_t OFF_PE = OFF_PF + A256(SZ_PF);

extern "C" void kernel_launch(void* const* d_in, const int* in_sizes, int n_in,
                              void* d_out, int out_size, void* d_ws, size_t ws_size,
                              hipStream_t stream) {
    (void)in_sizes; (void)n_in; (void)out_size; (void)ws_size;
    const float* x        = (const float*)d_in[0];   // (512,16,512)
    const float* pos_emb  = (const float*)d_in[1];   // (1,1023,192)
    const unsigned char* kpm = (const unsigned char*)d_in[2]; // (16,512) bool
    const float* W_in     = (const float*)d_in[3];   // (544,512)
    const float* b_in     = (const float*)d_in[4];   // (544,)
    const float* W_pos    = (const float*)d_in[5];   // (32,192)
    float* out            = (float*)d_out;           // (8,16,512,512)

    char* ws = (char*)d_ws;
    _Float16* xh = (_Float16*)(ws + OFF_XH);
    _Float16* wh = (_Float16*)(ws + OFF_WH);
    _Float16* qh = (_Float16*)(ws + OFF_QH);
    _Float16* kh = (_Float16*)(ws + OFF_KH);
    float*    pf = (float*)(ws + OFF_PF);
    float*    pe = (float*)(ws + OFF_PE);

    pe_kernel<<<dim3(RELLEN), dim3(32), 0, stream>>>(pos_emb, W_pos, pe);

    {
        const size_t total = (size_t)MROWS * EMB + (size_t)NPROJ * EMB;
        cvt_kernel<<<dim3((unsigned)((total + 255) / 256)), dim3(256), 0, stream>>>(
            x, W_in, xh, wh);
    }

    inproj_kernel<<<dim3(MROWS / 16 / 4, NPROJ / 16), dim3(128), 0, stream>>>(
        xh, wh, b_in, qh, kh, pf);

    attn_kernel<<<dim3(S_LEN / ROWS_PER_BLK, BATCH * HEADS), dim3(256), 0, stream>>>(
        qh, kh, pf, pe, kpm, out);
}
